// AttentionModule_29205777613747
// MI455X (gfx1250) — matrix operand order repaired and checked by reading, not compiled, not run on hardware
//
#include <hip/hip_runtime.h>
#include <hip/hip_bf16.h>
#include <stddef.h>

// ---------------------------------------------------------------------------
// Problem constants (from reference): B=8, H=W=64 -> n=4096, C=256, D=C/8=32
// ---------------------------------------------------------------------------
#define BB    8
#define NN    4096
#define CC    256
#define DD    32
#define MTOT  (BB * NN)          // 32768 rows across all batches

typedef __bf16 bf16;
typedef __attribute__((ext_vector_type(16))) __bf16 v16bf;
typedef __attribute__((ext_vector_type(8)))  float  v8f;
typedef __attribute__((ext_vector_type(4)))  unsigned int u32x4;
typedef __attribute__((ext_vector_type(2)))  unsigned int u32x2;
typedef __attribute__((ext_vector_type(2)))  int i32x2;
typedef __attribute__((ext_vector_type(4)))  int i32x4;

union Frag16 {                  // 16 bf16 per lane = one WMMA A/B fragment
    v16bf v;
    u32x4 u[2];
};

__device__ __forceinline__ v8f wmma_bf16(v16bf a, v16bf b, v8f c) {
    // D(f32 16x16) = A(bf16 16x32) * B(bf16 32x16) + C
    return __builtin_amdgcn_wmma_f32_16x16x32_bf16(
        /*neg_a=*/false, a, /*neg_b=*/false, b,
        /*c_mod=*/(short)0, c, /*reuse_a=*/false, /*reuse_b=*/false);
}

// ---------------------------------------------------------------------------
// CDNA5 async global->LDS copies (ASYNCcnt-tracked, no VGPR round-trip).
// Builtin params are typed vector pointers in addrspace(1)/(3):
//   b64 : (v2i as1*, v2i as3*, imm offset, imm cpol)
//   b128: (v4i as1*, v4i as3*, imm offset, imm cpol)
// Guarded so the file still compiles on toolchains without the builtins.
// ---------------------------------------------------------------------------
#if defined(__HIP_DEVICE_COMPILE__) &&                                        \
    __has_builtin(__builtin_amdgcn_global_load_async_to_lds_b128) &&          \
    __has_builtin(__builtin_amdgcn_global_load_async_to_lds_b64)
#define HAS_ASYNC_LDS 1
typedef __attribute__((address_space(1))) void GAS;
typedef __attribute__((address_space(3))) void LAS;
typedef __attribute__((address_space(1))) i32x2 g_i32x2;
typedef __attribute__((address_space(3))) i32x2 l_i32x2;
typedef __attribute__((address_space(1))) i32x4 g_i32x4;
typedef __attribute__((address_space(3))) i32x4 l_i32x4;
#if __has_builtin(__builtin_amdgcn_s_wait_asynccnt)
#define ASYNC_WAIT(n) __builtin_amdgcn_s_wait_asynccnt(n)
#else
#define ASYNC_WAIT(n) asm volatile("s_wait_asynccnt " #n ::: "memory")
#endif
#else
#define HAS_ASYNC_LDS 0
#define ASYNC_WAIT(n)
#endif

// ---------------------------------------------------------------------------
// Elementwise f32 -> bf16 convert (for x)
// ---------------------------------------------------------------------------
__global__ void cvt_f32_bf16(const float* __restrict__ in,
                             bf16* __restrict__ out, int n) {
    for (int i = blockIdx.x * blockDim.x + threadIdx.x; i < n;
         i += gridDim.x * blockDim.x)
        out[i] = (bf16)in[i];
}

// Wt[n][k] = (bf16) W[k][n]   (W row-major [K][N])
__global__ void transpose_w_bf16(const float* __restrict__ W,
                                 bf16* __restrict__ Wt, int K, int N) {
    int i = blockIdx.x * blockDim.x + threadIdx.x;
    if (i < N * K) {
        int n = i / K, k = i - n * K;
        Wt[i] = (bf16)W[(size_t)k * N + n];
    }
}

// ---------------------------------------------------------------------------
// GEMM: Y = alpha * (A[M,K](bf16) @ Bt[N,K]^T(bf16) + bias)
//   transY == 0 : Y row-major [M,N] (bf16)
//   transY == 1 : Y row-major [N,M] (bf16)  (transposed store; contiguous!)
// Block: 256 threads = 8 waves arranged 4 row-tiles x 2 col-tiles
// Grid:  (M/64, N/32)
// ---------------------------------------------------------------------------
__global__ __launch_bounds__(256)
void gemm_bf16(const bf16* __restrict__ A, const bf16* __restrict__ Bt,
               const float* __restrict__ bias, bf16* __restrict__ Y,
               int M, int N, int K, float alpha, int transY) {
    const int tid  = threadIdx.x;
    const int lane = tid & 31;
    const int wave = tid >> 5;
    const int half = lane >> 4;      // 0: lanes 0-15, 1: lanes 16-31
    const int l16  = lane & 15;

    const int m0 = blockIdx.x * 64 + (wave >> 1) * 16;
    const int n0 = blockIdx.y * 32 + (wave & 1) * 16;

    v8f acc = {0.f, 0.f, 0.f, 0.f, 0.f, 0.f, 0.f, 0.f};

    // A-fragment: lane row = m0+l16; K chunks {half*8 .. +7} and {+16 .. +23}
    const bf16* arow = A + (size_t)(m0 + l16) * K + half * 8;
    // B-fragment: lane col = n0+l16; K chunks {half*8 .. +7} and {+16 .. +23}
    const bf16* brow = Bt + (size_t)(n0 + l16) * K + half * 8;

    #pragma unroll
    for (int kk = 0; kk < CC; kk += 32) {   // K == 256 for all launches
        Frag16 a, b;
        a.u[0] = *(const u32x4*)(arow + kk);
        a.u[1] = *(const u32x4*)(arow + kk + 16);
        b.u[0] = *(const u32x4*)(brow + kk);
        b.u[1] = *(const u32x4*)(brow + kk + 16);
        acc = wmma_bf16(a.v, b.v, acc);
    }

    const int col = n0 + l16;               // D layout: lane -> column
    const float bv = bias[col];

    if (transY) {
        union { bf16 h[8]; u32x4 u; } pk;
        #pragma unroll
        for (int g = 0; g < 8; g++)
            pk.h[g] = (bf16)((acc[g] + bv) * alpha);
        // rows m0 + half*8 + (0..7) are consecutive -> one 16B store
        *(u32x4*)(Y + (size_t)col * M + m0 + half * 8) = pk.u;
    } else {
        #pragma unroll
        for (int g = 0; g < 8; g++) {
            int row = m0 + half * 8 + g;
            Y[(size_t)row * N + col] = (bf16)((acc[g] + bv) * alpha);
        }
    }
}

// ---------------------------------------------------------------------------
// Stage one 32-key block into LDS: K tile (32x32 bf16, 2KB) + Vt tile
// (256c x 32keys bf16, 16KB). 256 threads: 1x b64 + 4x b128 per thread.
// Async path: ASYNCcnt-tracked direct-to-LDS DMA, no VGPR round trip.
// ---------------------------------------------------------------------------
__device__ __forceinline__ void stage_tiles(const bf16* __restrict__ ksrc0,
                                            const bf16* __restrict__ vsrc0,
                                            bf16* kl, bf16* vl, int tid) {
    const bf16* ksrc = ksrc0 + (size_t)(tid >> 3) * DD + (tid & 7) * 4;
    bf16*       kdst = kl + (tid >> 3) * 32 + (tid & 7) * 4;
    const bf16* vsrc = vsrc0 + (size_t)tid * MTOT;
    bf16*       vdst = vl + tid * 32;
#if HAS_ASYNC_LDS
    __builtin_amdgcn_global_load_async_to_lds_b64(
        (g_i32x2*)(GAS*)ksrc, (l_i32x2*)(LAS*)kdst, 0, 0);
    #pragma unroll
    for (int j = 0; j < 4; j++)
        __builtin_amdgcn_global_load_async_to_lds_b128(
            (g_i32x4*)(GAS*)(vsrc + j * 8), (l_i32x4*)(LAS*)(vdst + j * 8),
            0, 0);
#else
    *(u32x2*)kdst = *(const u32x2*)ksrc;
    #pragma unroll
    for (int j = 0; j < 4; j++)
        ((u32x4*)vdst)[j] = ((const u32x4*)vsrc)[j];
#endif
}

// ---------------------------------------------------------------------------
// Flash-attention: per block = 128 query rows of one batch (8 waves x 16 rows)
// Streams 32-key blocks with double-buffered async LDS tiles; online softmax.
// q: [B*N, D] bf16 (pre-scaled by 1/sqrt(D));  k: [B*N, D] bf16
// vt: [C, B*N] bf16 (transposed V);  x,out: [B, N, C] f32
// Grid: (N/128, B), block 256.
// ---------------------------------------------------------------------------
__global__ __launch_bounds__(256)
void attn_flash(const bf16* __restrict__ q, const bf16* __restrict__ k,
                const bf16* __restrict__ vt, const float* __restrict__ x,
                const float* __restrict__ gamma_p, float* __restrict__ out) {
    __shared__ __align__(16) bf16 k_lds[2][32 * 32];     //  4 KB: [key][d]
    __shared__ __align__(16) bf16 v_lds[2][256 * 32];    // 32 KB: [c][key]
    __shared__ __align__(16) bf16 p_lds[8 * 16 * 32];    //  8 KB: per-wave P

    const int tid  = threadIdx.x;
    const int lane = tid & 31;
    const int wave = tid >> 5;
    const int half = lane >> 4;
    const int l16  = lane & 15;

    const int batch = blockIdx.y;
    const int qrow0 = blockIdx.x * 128 + wave * 16;      // within batch

    const bf16* qb  = q + (size_t)batch * NN * DD;
    const bf16* kb  = k + (size_t)batch * NN * DD;
    const bf16* vtb = vt + (size_t)batch * NN;           // row stride MTOT

    // q A-fragment (16 rows x 32 d) — one fragment, held for the whole loop
    Frag16 qa;
    {
        const bf16* base = qb + (size_t)(qrow0 + l16) * DD + half * 8;
        qa.u[0] = *(const u32x4*)(base);
        qa.u[1] = *(const u32x4*)(base + 16);
    }

    v8f acc[16];
    #pragma unroll
    for (int t = 0; t < 16; t++)
        #pragma unroll
        for (int g = 0; g < 8; g++) acc[t][g] = 0.f;

    float mrow[8], lrow[8];
    #pragma unroll
    for (int g = 0; g < 8; g++) { mrow[g] = -3.0e38f; lrow[g] = 0.f; }

    bf16* ps = &p_lds[wave * 16 * 32];

    const int T = NN / 32;                               // 128 key blocks
    stage_tiles(kb, vtb, k_lds[0], v_lds[0], tid);       // prologue: tile 0

    for (int it = 0; it < T; ++it) {
        const int buf = it & 1;
        // Issue next tile into the other buffer, then wait for the current
        // tile (<=5 in-flight loads belong to tile it+1 for this thread).
        if (it + 1 < T) {
            stage_tiles(kb + (size_t)(it + 1) * 32 * DD, vtb + (it + 1) * 32,
                        k_lds[buf ^ 1], v_lds[buf ^ 1], tid);
            ASYNC_WAIT(5);
        } else {
            ASYNC_WAIT(0);
        }
        __syncthreads();   // all threads' tile-it data visible in LDS

        const bf16* klt = k_lds[buf];
        const bf16* vlt = v_lds[buf];

        // --- scores: S = q (16x32) @ kT (32 x 16keys), two 16-key subtiles
        Frag16 b0, b1;
        {
            const bf16* kr0 = &klt[l16 * 32 + half * 8];
            b0.u[0] = *(const u32x4*)kr0;
            b0.u[1] = *(const u32x4*)(kr0 + 16);
            const bf16* kr1 = &klt[(16 + l16) * 32 + half * 8];
            b1.u[0] = *(const u32x4*)kr1;
            b1.u[1] = *(const u32x4*)(kr1 + 16);
        }
        v8f zero = {0.f, 0.f, 0.f, 0.f, 0.f, 0.f, 0.f, 0.f};
        v8f s0 = wmma_bf16(qa.v, b0.v, zero);
        v8f s1 = wmma_bf16(qa.v, b1.v, zero);

        // --- online softmax (rows live across the 16 lanes of each half)
        float corr[8];
        #pragma unroll
        for (int g = 0; g < 8; g++) {
            float mx = fmaxf(s0[g], s1[g]);
            #pragma unroll
            for (int s = 1; s < 16; s <<= 1)
                mx = fmaxf(mx, __shfl_xor(mx, s, 32));
            float mn = fmaxf(mrow[g], mx);
            float cf = __expf(mrow[g] - mn);
            float p0 = __expf(s0[g] - mn);
            float p1 = __expf(s1[g] - mn);
            s0[g] = p0; s1[g] = p1;
            float rs = p0 + p1;
            #pragma unroll
            for (int s = 1; s < 16; s <<= 1)
                rs += __shfl_xor(rs, s, 32);
            lrow[g] = lrow[g] * cf + rs;
            mrow[g] = mn;
            corr[g] = cf;
        }
        #pragma unroll
        for (int t = 0; t < 16; t++)
            #pragma unroll
            for (int g = 0; g < 8; g++) acc[t][g] *= corr[g];

        // --- transpose P (D-layout -> A-layout) through per-wave LDS scratch
        #pragma unroll
        for (int g = 0; g < 8; g++) {
            int row = half * 8 + g;
            ps[row * 32 + l16]      = (bf16)s0[g];
            ps[row * 32 + 16 + l16] = (bf16)s1[g];
        }
        Frag16 pa;   // wave-private scratch: lanes are lockstep, DS in-order
        {
            const bf16* base = ps + l16 * 32 + half * 8;
            pa.u[0] = *(const u32x4*)base;
            pa.u[1] = *(const u32x4*)(base + 16);
        }

        // --- PV: acc[t] += P (16x32keys) @ V (32keys x 16c), 16 column tiles
        #pragma unroll
        for (int t = 0; t < 16; t++) {
            Frag16 vb;
            const bf16* vr = &vlt[(t * 16 + l16) * 32 + half * 8];
            vb.u[0] = *(const u32x4*)vr;
            vb.u[1] = *(const u32x4*)(vr + 16);
            acc[t] = wmma_bf16(pa.v, vb.v, acc[t]);
        }
        __syncthreads();   // done reading buf before it is overwritten
    }

    // --- epilogue: out = gamma * (acc / l) + x   (streaming, non-temporal)
    const float gam = gamma_p[0];
    float rl[8];
    #pragma unroll
    for (int g = 0; g < 8; g++) rl[g] = 1.0f / lrow[g];
    const size_t obase = (size_t)batch * NN * CC;
    #pragma unroll
    for (int t = 0; t < 16; t++) {
        int col = t * 16 + l16;
        #pragma unroll
        for (int g = 0; g < 8; g++) {
            int row = qrow0 + half * 8 + g;
            size_t idx = obase + (size_t)row * CC + col;
            float xi  = __builtin_nontemporal_load(x + idx);
            float val = gam * (acc[t][g] * rl[g]) + xi;
            __builtin_nontemporal_store(val, out + idx);
        }
    }
}

// ---------------------------------------------------------------------------
// Host-side launch
// ---------------------------------------------------------------------------
extern "C" void kernel_launch(void* const* d_in, const int* in_sizes, int n_in,
                              void* d_out, int out_size, void* d_ws,
                              size_t ws_size, hipStream_t stream) {
    (void)in_sizes; (void)n_in; (void)out_size; (void)ws_size;

    const float* x     = (const float*)d_in[0];   // [8,64,64,256]
    const float* Wq    = (const float*)d_in[1];   // [256,32]
    const float* bq    = (const float*)d_in[2];   // [32]
    const float* Wk    = (const float*)d_in[3];   // [256,32]
    const float* bk    = (const float*)d_in[4];   // [32]
    const float* Wv    = (const float*)d_in[5];   // [256,256]
    const float* bv    = (const float*)d_in[6];   // [256]
    const float* gamma = (const float*)d_in[7];   // scalar

    // Workspace layout (bytes), all 16B-aligned:
    char* ws = (char*)d_ws;
    bf16* xbf = (bf16*)(ws);                               // 16,777,216
    bf16* qbf = (bf16*)(ws + 16777216);                    //  2,097,152
    bf16* kbf = (bf16*)(ws + 18874368);                    //  2,097,152
    bf16* vtb = (bf16*)(ws + 20971520);                    // 16,777,216
    bf16* wtq = (bf16*)(ws + 37748736);                    //     16,384
    bf16* wtk = wtq + 32 * 256;
    bf16* wtv = wtk + 32 * 256;                            //    131,072

    const float scale = 0.17677669529663687f;              // 1/sqrt(32)

    cvt_f32_bf16<<<2048, 256, 0, stream>>>(x, xbf, MTOT * CC);
    transpose_w_bf16<<<(32 * 256 + 255) / 256, 256, 0, stream>>>(Wq, wtq, CC, DD);
    transpose_w_bf16<<<(32 * 256 + 255) / 256, 256, 0, stream>>>(Wk, wtk, CC, DD);
    transpose_w_bf16<<<(256 * 256 + 255) / 256, 256, 0, stream>>>(Wv, wtv, CC, CC);

    // q = (x@Wq + bq) * scale ; k = x@Wk + bk ; Vt = (x@Wv + bv)^T
    gemm_bf16<<<dim3(MTOT / 64, 1), 256, 0, stream>>>(xbf, wtq, bq, qbf,
                                                      MTOT, DD, CC, scale, 0);
    gemm_bf16<<<dim3(MTOT / 64, 1), 256, 0, stream>>>(xbf, wtk, bk, kbf,
                                                      MTOT, DD, CC, 1.0f, 0);
    gemm_bf16<<<dim3(MTOT / 64, CC / 32), 256, 0, stream>>>(xbf, wtv, bv, vtb,
                                                            MTOT, CC, CC, 1.0f, 1);

    attn_flash<<<dim3(NN / 128, BB), 256, 0, stream>>>(qbf, kbf, vtb, x, gamma,
                                                       (float*)d_out);
}
